// DPOTNet3D_49435073577151
// MI455X (gfx1250) — matrix operand order repaired and checked by reading, not compiled, not run on hardware
//
#include <hip/hip_runtime.h>
#include <hip/hip_bf16.h>
#include <math.h>

typedef __attribute__((ext_vector_type(16))) _Float16 v16h;
typedef __attribute__((ext_vector_type(8)))  float    v8f;

#define B_   2
#define C_   128
#define NX   64
#define NY   64
#define NZ   64
#define KX   32
#define KY   32
#define KZ   8
#define NB   8
#define TWO_PI 6.283185307179586f
#define ORTHO 0.125f   /* 1/sqrt(64), one per transformed axis (norm='ortho') */

// ---------------------------------------------------------------------------
// CDNA5 async global->LDS copy helpers (tracked by ASYNCcnt).
// LDS dest offset lds_off, global src sbase + voff   (GVS mode, §4.4)
// ---------------------------------------------------------------------------
__device__ __forceinline__ void async_b128(unsigned lds_off, unsigned voff,
                                           const void* sbase) {
  asm volatile("global_load_async_to_lds_b128 %0, %1, %2"
               :: "v"(lds_off), "v"(voff), "s"(sbase) : "memory");
}
__device__ __forceinline__ void async_b64(unsigned lds_off, unsigned voff,
                                          const void* sbase) {
  asm volatile("global_load_async_to_lds_b64 %0, %1, %2"
               :: "v"(lds_off), "v"(voff), "s"(sbase) : "memory");
}
__device__ __forceinline__ void async_wait0() {
  asm volatile("s_wait_asynccnt 0x0" ::: "memory");
}
// Low 32 bits of a generic pointer to LDS == wave-relative LDS byte offset
// (CDNA5 aperture decode, ISA §10.2).
__device__ __forceinline__ unsigned lds_off_of(const void* p) {
  return (unsigned)(size_t)p;
}

// ---------------------------------------------------------------------------
// Stage 1: real->complex DFT along Z, keep only bins 0..7.  One block per
// (b,c,x) plane: async-stages the 64x64 plane, computes 64 pencils x 8 bins.
// LDS row stride 68 floats = 272 B (16B-aligned for b128, bank-conflict free).
// ---------------------------------------------------------------------------
__global__ __launch_bounds__(256) void k_fwd_z(const float* __restrict__ xin,
                                               float2* __restrict__ outA) {
  __shared__ float tile[NY][NZ + 4];
  __shared__ float twc[NZ], tws[NZ];
  const int t = threadIdx.x;
  const int plane = blockIdx.x;                       // (b*C + c)*NX + x
  const float* src = xin + (size_t)plane * (NY * NZ);
  // async-stage 16 KB (1024 x b128); lds has padded rows
  {
    const unsigned base = lds_off_of(&tile[0][0]);
    #pragma unroll
    for (int i = 0; i < 4; ++i) {
      const unsigned q = t + 256 * i;                 // 16-byte chunk id
      const unsigned y = q >> 4, zq = q & 15;
      async_b128(base + y * ((NZ + 4) * 4) + zq * 16, q * 16, src);
    }
  }
  if (t < NZ) {
    float s, c;
    sincosf(TWO_PI * (float)t / (float)NZ, &s, &c);
    twc[t] = c; tws[t] = -s;                          // e^{-i theta}
  }
  async_wait0();
  __syncthreads();
  float2* dst = outA + (size_t)plane * (NY * KZ);
  for (int o = t; o < NY * KZ; o += 256) {
    const int y = o >> 3, k = o & 7;
    float re = 0.f, im = 0.f;
    #pragma unroll 8
    for (int z = 0; z < NZ; ++z) {
      const float v = tile[y][z];
      const int j = (k * z) & 63;
      re = fmaf(v, twc[j], re);
      im = fmaf(v, tws[j], im);
    }
    dst[o] = make_float2(re * ORTHO, im * ORTHO);
  }
}

// ---------------------------------------------------------------------------
// Stage 2: complex DFT along Y, keep bins 0..31.  Block per (b,c,x).
// 4 KB contiguous tile -> one async b128 per thread.
// ---------------------------------------------------------------------------
__global__ __launch_bounds__(256) void k_fwd_y(const float2* __restrict__ inA,
                                               float2* __restrict__ outB) {
  __shared__ float2 tile[NY][KZ];
  __shared__ float twc[NY], tws[NY];
  const int t = threadIdx.x;
  const int plane = blockIdx.x;                       // (b*C + c)*NX + x
  const float2* src = inA + (size_t)plane * (NY * KZ);
  async_b128(lds_off_of(&tile[0][0]) + t * 16, t * 16, src);
  if (t < NY) { float s, c; sincosf(TWO_PI * t / 64.f, &s, &c); twc[t] = c; tws[t] = -s; }
  async_wait0();
  __syncthreads();
  const int ky = t >> 3, kz = t & 7;                  // 256 outputs exactly
  float re = 0.f, im = 0.f;
  #pragma unroll 8
  for (int y = 0; y < NY; ++y) {
    const float2 v = tile[y][kz];
    const int j = (ky * y) & 63;
    const float c = twc[j], s = tws[j];
    re += v.x * c - v.y * s;
    im += v.x * s + v.y * c;
  }
  outB[(size_t)plane * (KY * KZ) + t] = make_float2(re * ORTHO, im * ORTHO);
}

// ---------------------------------------------------------------------------
// Stage 3: complex DFT along X, keep bins 0..31; write channel-last "sel"
// sel[m*C + c], m = ((b*KX+kx)*KY+ky)*KZ+kz.   Block per (b,c,ky).
// Strided 4 KB tile (64 rows of 8 float2) -> one async b128 per thread.
// ---------------------------------------------------------------------------
__global__ __launch_bounds__(256) void k_fwd_x(const float2* __restrict__ inB,
                                               float2* __restrict__ sel) {
  __shared__ float2 tile[NX][KZ];
  __shared__ float twc[NX], tws[NX];
  const int t = threadIdx.x;
  const int blk = blockIdx.x;                         // b*C*KY + c*KY + ky
  const int ky = blk & 31;
  const int c  = (blk >> 5) & 127;
  const int b  = blk >> 12;
  const int bc = b * C_ + c;
  {
    // chunk t: row xx = t>>2, pair p = t&3 covers kz = 2p, 2p+1
    const float2* fb = inB + ((size_t)bc * NX * KY + ky) * KZ;  // (xx=0,kz=0)
    const unsigned xx = t >> 2, p = t & 3;
    async_b128(lds_off_of(&tile[0][0]) + xx * 64 + p * 16,
               xx * (KY * KZ * 8) + p * 16, fb);
  }
  if (t < NX) { float s, cc; sincosf(TWO_PI * t / 64.f, &s, &cc); twc[t] = cc; tws[t] = -s; }
  async_wait0();
  __syncthreads();
  const int kx = t >> 3, kz = t & 7;
  float re = 0.f, im = 0.f;
  #pragma unroll 8
  for (int x = 0; x < NX; ++x) {
    const float2 v = tile[x][kz];
    const int j = (kx * x) & 63;
    const float cc = twc[j], s = tws[j];
    re += v.x * cc - v.y * s;
    im += v.x * s + v.y * cc;
  }
  const size_t m = (((size_t)b * KX + kx) * KY + ky) * KZ + kz;
  sel[m * C_ + c] = make_float2(re * ORTHO, im * ORTHO);
}

// ---------------------------------------------------------------------------
// Mode mixing: complex block-diagonal 2-layer MLP via WMMA.
// One workgroup = 16 modes (rows); wave n (of 8) = channel block n.
// Complex GEMM folded into K=32:  A = [re | im],  B = [Wr ; -Wi] / [Wi ; Wr].
// 4 x v_wmma_f32_16x16x32_f16 per wave (layer1 re/im, layer2 re/im).
// VGPR layouts per CDNA5 ISA 7.12.2 (wave32).  Tile staged async (16 KB).
// ---------------------------------------------------------------------------
__global__ __launch_bounds__(256) void k_mix(const float2* __restrict__ sel,
                                             float2* __restrict__ mixed,
                                             const float* __restrict__ w1,
                                             const float* __restrict__ b1,
                                             const float* __restrict__ w2,
                                             const float* __restrict__ b2) {
  __shared__ float2 tile[16][C_];       // 16 modes x 128 channels (16 KB)
  __shared__ float  hs[2][NB][16][16];  // hidden re/im per block (16 KB)
  const int t    = threadIdx.x;
  const int n    = t >> 5;              // wave id == channel block
  const int lane = t & 31;
  const size_t row0 = (size_t)blockIdx.x * 16;

  {
    const float2* src = sel + row0 * C_;              // contiguous 16 KB
    const unsigned base = lds_off_of(&tile[0][0]);
    #pragma unroll
    for (int i = 0; i < 4; ++i) {
      const unsigned q = t + 256 * i;
      async_b128(base + q * 16, q * 16, src);
    }
  }
  async_wait0();
  __syncthreads();

  const float* w1r = w1 + n * 256;
  const float* w1i = w1 + 2048 + n * 256;
  const float* w2r = w2 + n * 256;
  const float* w2i = w2 + 2048 + n * 256;
  const int col  = lane & 15;           // N (output channel in block)
  const int hi   = lane >> 4;           // lane half selects K sub-range
  const int m    = lane & 15;           // A row held by this lane
  const int iOff = hi << 3;             // input-channel offset 0 or 8

  // ---- layer 1 operands --------------------------------------------------
  // A (16x32 f16): lanes 0-15: e0..7 = K0..7, e8..15 = K16..23
  //                lanes16-31: e0..7 = K8..15, e8..15 = K24..31
  v16h a, br, bi;
  #pragma unroll
  for (int j = 0; j < 8; ++j) {
    const float2 v = tile[m][n * 16 + iOff + j];
    a[j]     = (_Float16)v.x;           // K 0..15  <- real part
    a[8 + j] = (_Float16)v.y;           // K 16..31 <- imag part
  }
  // B (32x16 f16): same K layout as A (e0..7 = K iOff.., e8..15 = K 16+iOff..)
  #pragma unroll
  for (int j = 0; j < 16; ++j) {
    const float wr = w1r[(iOff + (j & 7)) * 16 + col];
    const float wi = w1i[(iOff + (j & 7)) * 16 + col];
    br[j] = (_Float16)((j >= 8) ? -wi : wr);  // [W1r ; -W1i]
    bi[j] = (_Float16)((j >= 8) ?  wr : wi);  // [W1i ;  W1r]
  }
  v8f accR, accI;
  const float b1r  = b1[n * 16 + col];
  const float b1iv = b1[C_ + n * 16 + col];
  #pragma unroll
  for (int r = 0; r < 8; ++r) { accR[r] = b1r; accI[r] = b1iv; }
  accR = __builtin_amdgcn_wmma_f32_16x16x32_f16(false, a, false, br, (short)0, accR, false, false);
  accI = __builtin_amdgcn_wmma_f32_16x16x32_f16(false, a, false, bi, (short)0, accI, false, false);

  // exact GELU: 0.5*x*(1+erf(x/sqrt(2)))
  #pragma unroll
  for (int r = 0; r < 8; ++r) {
    const float xr = accR[r]; accR[r] = 0.5f * xr * (1.f + erff(xr * 0.70710678118654752f));
    const float xi = accI[r]; accI[r] = 0.5f * xi * (1.f + erff(xi * 0.70710678118654752f));
  }
  // C/D layout: VGPR r -> row (r + 8*hi), col = lane&15
  #pragma unroll
  for (int r = 0; r < 8; ++r) {
    const int row = r + (hi << 3);
    hs[0][n][row][col] = accR[r];
    hs[1][n][row][col] = accI[r];
  }
  __syncthreads();

  // ---- layer 2 -----------------------------------------------------------
  v16h a2, br2, bi2;
  #pragma unroll
  for (int j = 0; j < 8; ++j) {
    a2[j]     = (_Float16)hs[0][n][m][iOff + j];
    a2[8 + j] = (_Float16)hs[1][n][m][iOff + j];
  }
  #pragma unroll
  for (int j = 0; j < 16; ++j) {
    const float wr = w2r[(iOff + (j & 7)) * 16 + col];
    const float wi = w2i[(iOff + (j & 7)) * 16 + col];
    br2[j] = (_Float16)((j >= 8) ? -wi : wr);
    bi2[j] = (_Float16)((j >= 8) ?  wr : wi);
  }
  v8f oR, oI;
  const float b2r  = b2[n * 16 + col];
  const float b2iv = b2[C_ + n * 16 + col];
  #pragma unroll
  for (int r = 0; r < 8; ++r) { oR[r] = b2r; oI[r] = b2iv; }
  oR = __builtin_amdgcn_wmma_f32_16x16x32_f16(false, a2, false, br2, (short)0, oR, false, false);
  oI = __builtin_amdgcn_wmma_f32_16x16x32_f16(false, a2, false, bi2, (short)0, oI, false, false);

  #pragma unroll
  for (int r = 0; r < 8; ++r) {
    const int row = r + (hi << 3);
    mixed[(row0 + row) * C_ + n * 16 + col] = make_float2(oR[r], oI[r]);
  }
}

// ---------------------------------------------------------------------------
// Inverse X: zero-padded inverse DFT (sum over kx<32, e^{+i}).  Block per
// (b,ky,kz), 128 threads = channels.  32 async b64 per thread stage 32 KB.
// ---------------------------------------------------------------------------
__global__ __launch_bounds__(128) void k_inv_x(const float2* __restrict__ mixed,
                                               float2* __restrict__ ox) {
  __shared__ float2 lb[KX][C_];         // 32 KB
  __shared__ float twc[64], tws[64];
  const int t = threadIdx.x;            // channel
  const int blk = blockIdx.x;           // b*KY*KZ + ky*KZ + kz
  const int kz = blk & 7, ky = (blk >> 3) & 31, b = blk >> 8;
  {
    const float2* sb = mixed + (((size_t)b * KX * KY + ky) * KZ + kz) * C_; // kx=0
    const unsigned base = lds_off_of(&lb[0][0]);
    #pragma unroll 8
    for (int kx = 0; kx < KX; ++kx)
      async_b64(base + (kx * C_ + t) * 8, kx * (KY * KZ * C_ * 8) + t * 8, sb);
  }
  if (t < 64) { float s, c; sincosf(TWO_PI * t / 64.f, &s, &c); twc[t] = c; tws[t] = s; }
  async_wait0();
  __syncthreads();
  for (int x = 0; x < NX; ++x) {
    float re = 0.f, im = 0.f;
    #pragma unroll 8
    for (int kx = 0; kx < KX; ++kx) {
      const float2 v = lb[kx][t];
      const int j = (kx * x) & 63;
      const float c = twc[j], s = tws[j];
      re += v.x * c - v.y * s;
      im += v.x * s + v.y * c;
    }
    ox[((((size_t)b * NX + x) * KY + ky) * KZ + kz) * C_ + t] =
        make_float2(re * ORTHO, im * ORTHO);
  }
}

// ---------------------------------------------------------------------------
// Inverse Y: same structure.  Block per (b,x,kz).
// ---------------------------------------------------------------------------
__global__ __launch_bounds__(128) void k_inv_y(const float2* __restrict__ ox,
                                               float2* __restrict__ oy) {
  __shared__ float2 lb[KY][C_];         // 32 KB
  __shared__ float twc[64], tws[64];
  const int t = threadIdx.x;
  const int blk = blockIdx.x;           // b*NX*KZ + x*KZ + kz
  const int kz = blk & 7, x = (blk >> 3) & 63, b = blk >> 9;
  {
    const float2* sb = ox + (((size_t)b * NX + x) * KY * KZ + kz) * C_;     // ky=0
    const unsigned base = lds_off_of(&lb[0][0]);
    #pragma unroll 8
    for (int ky = 0; ky < KY; ++ky)
      async_b64(base + (ky * C_ + t) * 8, ky * (KZ * C_ * 8) + t * 8, sb);
  }
  if (t < 64) { float s, c; sincosf(TWO_PI * t / 64.f, &s, &c); twc[t] = c; tws[t] = s; }
  async_wait0();
  __syncthreads();
  for (int y = 0; y < NY; ++y) {
    float re = 0.f, im = 0.f;
    #pragma unroll 8
    for (int ky = 0; ky < KY; ++ky) {
      const float2 v = lb[ky][t];
      const int j = (ky * y) & 63;
      const float c = twc[j], s = tws[j];
      re += v.x * c - v.y * s;
      im += v.x * s + v.y * c;
    }
    oy[((((size_t)b * NX + x) * NY + y) * KZ + kz) * C_ + t] =
        make_float2(re * ORTHO, im * ORTHO);
  }
}

// ---------------------------------------------------------------------------
// Inverse Z (irfft, hermitian): out[z] = (1/8)*(X0.re + 2*sum_{k=1..7}
// (Xk.re cos - Xk.im sin)), then add residual and transpose to (B,C,X,Y,Z).
// Block per (b,x,y); 8 KB contiguous tile -> 2 async b128 per thread.
// ---------------------------------------------------------------------------
__global__ __launch_bounds__(256) void k_inv_z(const float2* __restrict__ oy,
                                               const float* __restrict__ xin,
                                               float* __restrict__ out) {
  __shared__ float2 lz[KZ][C_];         // 8 KB
  __shared__ float twc[64], tws[64];
  const int t = threadIdx.x;
  const int blk = blockIdx.x;           // b*NX*NY + x*NY + y
  const int y = blk & 63, x = (blk >> 6) & 63, b = blk >> 12;
  {
    const float2* src = oy + (((size_t)b * NX + x) * NY + y) * KZ * C_;     // contiguous
    const unsigned base = lds_off_of(&lz[0][0]);
    async_b128(base + t * 16, t * 16, src);
    async_b128(base + (t + 256) * 16, (t + 256) * 16, src);
  }
  if (t < 64) { float s, c; sincosf(TWO_PI * t / 64.f, &s, &c); twc[t] = c; tws[t] = s; }
  async_wait0();
  __syncthreads();
  const int z = t & 63;
  for (int ci = t >> 6; ci < C_; ci += 4) {
    float acc = lz[0][ci].x;
    #pragma unroll
    for (int k = 1; k < KZ; ++k) {
      const int j = (k * z) & 63;
      const float2 v = lz[k][ci];
      acc += 2.f * (v.x * twc[j] - v.y * tws[j]);
    }
    acc *= ORTHO;
    const size_t oi = ((((size_t)b * C_ + ci) * NX + x) * NY + y) * NZ + z;
    out[oi] = acc + xin[oi];
  }
}

// ---------------------------------------------------------------------------
extern "C" void kernel_launch(void* const* d_in, const int* in_sizes, int n_in,
                              void* d_out, int out_size, void* d_ws, size_t ws_size,
                              hipStream_t stream) {
  (void)in_sizes; (void)n_in; (void)out_size; (void)ws_size;
  const float* x  = (const float*)d_in[0];
  const float* w1 = (const float*)d_in[1];
  const float* b1 = (const float*)d_in[2];
  const float* w2 = (const float*)d_in[3];
  const float* b2 = (const float*)d_in[4];
  float* out = (float*)d_out;

  char* ws = (char*)d_ws;
  float2* bufA  = (float2*)(ws);                                    // 64 MiB: B*C*NX*NY*KZ
  float2* bufB  = (float2*)(ws + (size_t)67108864);                 // 32 MiB: B*C*NX*KY*KZ
  float2* sel   = (float2*)(ws + (size_t)67108864 + 33554432);      // 16 MiB
  float2* mixed = (float2*)(ws + (size_t)67108864 + 33554432 + 16777216); // 16 MiB
  float2* ox = bufB;   // reuse: bufB free after k_fwd_x
  float2* oy = bufA;   // reuse: bufA free after k_fwd_y

  k_fwd_z<<<B_ * C_ * NX, 256, 0, stream>>>(x, bufA);
  k_fwd_y<<<B_ * C_ * NX, 256, 0, stream>>>(bufA, bufB);
  k_fwd_x<<<B_ * C_ * KY, 256, 0, stream>>>(bufB, sel);
  k_mix  <<<(B_ * KX * KY * KZ) / 16, 256, 0, stream>>>(sel, mixed, w1, b1, w2, b2);
  k_inv_x<<<B_ * KY * KZ, 128, 0, stream>>>(mixed, ox);
  k_inv_y<<<B_ * NX * KZ, 128, 0, stream>>>(ox, oy);
  k_inv_z<<<B_ * NX * NY, 256, 0, stream>>>(oy, x, out);
}
